// PointNet2Encoder_67147518705714
// MI455X (gfx1250) — matrix operand order repaired and checked by reading, not compiled, not run on hardware
//
#include <hip/hip_runtime.h>

typedef __attribute__((ext_vector_type(16))) _Float16 v16h;
typedef __attribute__((ext_vector_type(8)))  _Float16 v8h;
typedef __attribute__((ext_vector_type(8)))  float    v8f;

// ---------------------------------------------------------------------------
// Furthest point sampling: one block per batch. dists kept in LDS, argmax with
// first-occurrence tie-break (matches jnp.argmax). Emits new_xyz[b,i,:].
// ---------------------------------------------------------------------------
#define FPS_THREADS 256
__global__ __launch_bounds__(FPS_THREADS) void fps_kernel(
    const float* __restrict__ xyz, int N, int npoint, float* __restrict__ out_xyz)
{
    __shared__ float dist[4096];
    __shared__ float rv[FPS_THREADS];
    __shared__ int   ri[FPS_THREADS];
    const int b = blockIdx.x;
    const int t = threadIdx.x;
    const float* px = xyz + (size_t)b * N * 3;
    for (int n = t; n < N; n += FPS_THREADS) dist[n] = 1e10f;
    __syncthreads();
    int far = 0;
    for (int i = 0; i < npoint; ++i) {
        float cx = px[3 * far], cy = px[3 * far + 1], cz = px[3 * far + 2];
        if (t == 0) {
            float* o = out_xyz + ((size_t)b * npoint + i) * 3;
            o[0] = cx; o[1] = cy; o[2] = cz;
        }
        float bv = -1.0f; int bi = 0x7fffffff;
        for (int n = t; n < N; n += FPS_THREADS) {
            float dx = px[3 * n] - cx, dy = px[3 * n + 1] - cy, dz = px[3 * n + 2] - cz;
            float d  = dx * dx + dy * dy + dz * dz;
            float dn = fminf(dist[n], d);
            dist[n] = dn;
            if (dn > bv || (dn == bv && n < bi)) { bv = dn; bi = n; }
        }
        rv[t] = bv; ri[t] = bi;
        __syncthreads();
        for (int st = FPS_THREADS / 2; st > 0; st >>= 1) {
            if (t < st) {
                if (rv[t + st] > rv[t] || (rv[t + st] == rv[t] && ri[t + st] < ri[t])) {
                    rv[t] = rv[t + st]; ri[t] = ri[t + st];
                }
            }
            __syncthreads();
        }
        far = ri[0];
        __syncthreads();
    }
}

// ---------------------------------------------------------------------------
// Ball query (per batch): one thread per center, first <=K in-ball points in
// index order, padded with the first hit. out[s*K + j].
// ---------------------------------------------------------------------------
__global__ void ball_query_kernel(const float* __restrict__ xyz,
                                  const float* __restrict__ centers,
                                  int N, int S, int K, float r2, int* __restrict__ out)
{
    int s = blockIdx.x * blockDim.x + threadIdx.x;
    if (s >= S) return;
    float cx = centers[3 * s], cy = centers[3 * s + 1], cz = centers[3 * s + 2];
    int* row = out + (size_t)s * K;
    int cnt = 0, first = 0;
    for (int n = 0; n < N; ++n) {
        float dx = xyz[3 * n] - cx, dy = xyz[3 * n + 1] - cy, dz = xyz[3 * n + 2] - cz;
        float d2 = dx * dx + dy * dy + dz * dz;
        if (d2 < r2) {
            if (cnt == 0) first = n;
            row[cnt++] = n;
            if (cnt == K) break;
        }
    }
    for (int j = cnt; j < K; ++j) row[j] = first;
}

// ---------------------------------------------------------------------------
// Build grouped input matrix (per batch): X[(s*K+k)*ldx + c] in f16,
// c<3: (centered) xyz, c<3+C: features (f16 or f32 source), else zero pad.
// idx==nullptr -> identity grouping (group-all), pt = k.
// ---------------------------------------------------------------------------
__global__ void build_grouped_kernel(const float* __restrict__ xyz,
                                     const float* __restrict__ centers,
                                     const float* __restrict__ featsF,
                                     const _Float16* __restrict__ featsH,
                                     const int* __restrict__ idx,
                                     _Float16* __restrict__ X,
                                     int S, int K, int C, int ldx, int subtract)
{
    size_t e = (size_t)blockIdx.x * blockDim.x + threadIdx.x;
    size_t total = (size_t)S * K * ldx;
    if (e >= total) return;
    int c = (int)(e % ldx);
    size_t row = e / ldx;
    int k = (int)(row % K);
    int s = (int)(row / K);
    int pt = idx ? idx[row] : k;
    float v = 0.0f;
    if (c < 3) {
        v = xyz[(size_t)pt * 3 + c];
        if (subtract) v -= centers[(size_t)s * 3 + c];
    } else if (c < 3 + C) {
        v = featsH ? (float)featsH[(size_t)pt * C + (c - 3)]
                   : featsF[(size_t)pt * 3 + (c - 3)];
    }
    X[e] = (_Float16)v;
}

// ---------------------------------------------------------------------------
// Weight convert: f32 [Cout,Cin] -> f16 [Cout,ldw] zero-padded.
// ---------------------------------------------------------------------------
__global__ void convert_w_kernel(const float* __restrict__ W, _Float16* __restrict__ Wh,
                                 int Cout, int Cin, int ldw)
{
    int t = blockIdx.x * blockDim.x + threadIdx.x;
    if (t >= Cout * ldw) return;
    int o = t / ldw, c = t % ldw;
    Wh[t] = (_Float16)(c < Cin ? W[(size_t)o * Cin + c] : 0.0f);
}

// ---------------------------------------------------------------------------
// WMMA GEMM + ReLU: Y[R,Cout](f16) = relu( X[R,Kpad](f16) * W[Cout,Kpad]^T ),
// f32 accumulate via v_wmma_f32_16x16x32_f16. One wave computes a 16x32 tile
// (two N-accumulators share one A fragment -> 2 wmma per 3 fragment loads).
// A frag: lane l -> row M = l&15; K = hi*16 .. hi*16+15 (contiguous).
// B frag: lane l -> col N = l&15; K = hi*16 .. hi*16+15 (contiguous).
// C/D   : lane l -> col N = l&15; vgpr r -> row M = r + hi*8.
// Requires Cout % 32 == 0 (true for 32/64/96/128/512).
// ---------------------------------------------------------------------------
__global__ __launch_bounds__(128) void gemm_relu_wmma_kernel(
    const _Float16* __restrict__ X, const _Float16* __restrict__ W,
    _Float16* __restrict__ Y, int R, int Kpad, int Cout)
{
    const int ntN   = Cout >> 5;               // N-tiles of 32
    const int tiles = (R >> 4) * ntN;
    int wave = blockIdx.x * (blockDim.x >> 5) + (threadIdx.x >> 5);
    if (wave >= tiles) return;                 // wave-uniform: EXEC stays all-ones
    int tM = wave / ntN, tN = wave % ntN;
    int lane = threadIdx.x & 31;
    int mr = lane & 15, hi = lane >> 4;
    const _Float16* xrow  = X + (size_t)(tM * 16 + mr) * Kpad;
    const _Float16* wrow0 = W + (size_t)(tN * 32 + mr) * Kpad;
    const _Float16* wrow1 = wrow0 + (size_t)16 * Kpad;
    v8f acc0 = {};
    v8f acc1 = {};
    for (int k0 = 0; k0 < Kpad; k0 += 32) {
        v8h a0 = *(const v8h*)(xrow + k0 + hi * 16);
        v8h a1 = *(const v8h*)(xrow + k0 + hi * 16 + 8);
        v16h b0 = *(const v16h*)(wrow0 + k0 + hi * 16);
        v16h b1 = *(const v16h*)(wrow1 + k0 + hi * 16);
        v16h a;
#pragma unroll
        for (int j = 0; j < 8; ++j) { a[j] = a0[j]; a[j + 8] = a1[j]; }
        acc0 = __builtin_amdgcn_wmma_f32_16x16x32_f16(
            false, a, false, b0, (short)0, acc0, false, false);
        acc1 = __builtin_amdgcn_wmma_f32_16x16x32_f16(
            false, a, false, b1, (short)0, acc1, false, false);
    }
    _Float16* yb = Y + (size_t)(tM * 16 + hi * 8) * Cout + tN * 32 + mr;
#pragma unroll
    for (int r = 0; r < 8; ++r) {
        float v0 = acc0[r];
        float v1 = acc1[r];
        yb[(size_t)r * Cout]      = (_Float16)(v0 > 0.0f ? v0 : 0.0f);
        yb[(size_t)r * Cout + 16] = (_Float16)(v1 > 0.0f ? v1 : 0.0f);
    }
}

// ---------------------------------------------------------------------------
// Segmented max over K rows per group. f16->f16 (stage pooled feats) and
// f16->f32 (final output) variants.
// ---------------------------------------------------------------------------
__global__ void maxpool_h_kernel(const _Float16* __restrict__ Y, int G, int K, int C,
                                 _Float16* __restrict__ out, int ldo)
{
    int t = blockIdx.x * blockDim.x + threadIdx.x;
    if (t >= G * C) return;
    int g = t / C, c = t % C;
    const _Float16* p = Y + (size_t)g * K * C + c;
    float m = -3.0e38f;
    for (int j = 0; j < K; ++j) { float v = (float)p[(size_t)j * C]; m = v > m ? v : m; }
    out[(size_t)g * ldo + c] = (_Float16)m;
}

__global__ void maxpool_f_kernel(const _Float16* __restrict__ Y, int G, int K, int C,
                                 float* __restrict__ out)
{
    int t = blockIdx.x * blockDim.x + threadIdx.x;
    if (t >= G * C) return;
    int g = t / C, c = t % C;
    const _Float16* p = Y + (size_t)g * K * C + c;
    float m = -3.0e38f;
    for (int j = 0; j < K; ++j) { float v = (float)p[(size_t)j * C]; m = v > m ? v : m; }
    out[(size_t)g * C + c] = m;
}

// ---------------------------------------------------------------------------
// Host orchestration
// ---------------------------------------------------------------------------
extern "C" void kernel_launch(void* const* d_in, const int* in_sizes, int n_in,
                              void* d_out, int out_size, void* d_ws, size_t ws_size,
                              hipStream_t stream)
{
    (void)in_sizes; (void)n_in; (void)out_size; (void)ws_size;
    const float* x = (const float*)d_in[0];
    const int B = 8, N1 = 4096, S1 = 512, S2 = 128;

    struct WSpec { int cout, cin, ldw; };
    const WSpec wspec[19] = {
        {32, 6, 32}, {64, 32, 32},                       // sa1 scale0
        {64, 6, 32}, {64, 64, 64}, {128, 64, 64},        // sa1 scale1
        {64, 6, 32}, {96, 64, 64}, {128, 96, 96},        // sa1 scale2
        {64, 323, 352}, {64, 64, 64}, {128, 64, 64},     // sa2 scale0
        {64, 323, 352}, {64, 64, 64}, {128, 64, 64},     // sa2 scale1
        {64, 323, 352}, {64, 64, 64}, {128, 64, 64},     // sa2 scale2
        {512, 387, 416}, {512, 512, 512}                 // sa3
    };

    char* p = (char*)d_ws;
    auto alloc = [&](size_t bytes) -> char* {
        char* r = p; p += (bytes + 255) & ~(size_t)255; return r;
    };

    _Float16* wh[19];
    for (int i = 0; i < 19; ++i)
        wh[i] = (_Float16*)alloc((size_t)wspec[i].cout * wspec[i].ldw * 2);
    float*     nx1     = (float*)alloc((size_t)B * S1 * 3 * 4);
    float*     nx2     = (float*)alloc((size_t)B * S2 * 3 * 4);
    _Float16*  pooled1 = (_Float16*)alloc((size_t)B * S1 * 320 * 2);
    _Float16*  pooled2 = (_Float16*)alloc((size_t)B * S2 * 384 * 2);
    int*       bidx    = (int*)alloc((size_t)S1 * 128 * 4);          // per-batch ball idx
    _Float16*  bufX    = (_Float16*)alloc((size_t)16384 * 352 * 2);  // max grouped X
    _Float16*  bufA    = (_Float16*)alloc((size_t)65536 * 128 * 2);  // activation ping
    _Float16*  bufB    = (_Float16*)alloc((size_t)65536 * 128 * 2);  // activation pong

    // --- convert all weights to padded f16 ---
    for (int i = 0; i < 19; ++i) {
        int tot = wspec[i].cout * wspec[i].ldw;
        convert_w_kernel<<<(tot + 255) / 256, 256, 0, stream>>>(
            (const float*)d_in[i + 1], wh[i], wspec[i].cout, wspec[i].cin, wspec[i].ldw);
    }

    // --- SA1 ---
    fps_kernel<<<B, FPS_THREADS, 0, stream>>>(x, N1, S1, nx1);
    {
        const float radii[3]  = {0.1f, 0.2f, 0.4f};
        const int   ks[3]     = {16, 32, 128};
        const int   nlay[3]   = {2, 3, 3};
        const int   wbase[3]  = {0, 2, 5};
        int coff = 0;
        for (int sc = 0; sc < 3; ++sc) {
            int K = ks[sc]; float r2 = radii[sc] * radii[sc];
            for (int b = 0; b < B; ++b) {
                const float* xb = x   + (size_t)b * N1 * 3;
                const float* cb = nx1 + (size_t)b * S1 * 3;
                ball_query_kernel<<<(S1 + 127) / 128, 128, 0, stream>>>(
                    xb, cb, N1, S1, K, r2, bidx);
                int R = S1 * K, ldx = 32;
                size_t tot = (size_t)R * ldx;
                build_grouped_kernel<<<(int)((tot + 255) / 256), 256, 0, stream>>>(
                    xb, cb, xb, (const _Float16*)nullptr, bidx, bufX, S1, K, 3, ldx, 1);
                const _Float16* cur = bufX; int curLd = ldx;
                _Float16* nxt = bufA;
                for (int l = 0; l < nlay[sc]; ++l) {
                    const WSpec& w = wspec[wbase[sc] + l];
                    int tiles = (R / 16) * (w.cout / 32);
                    gemm_relu_wmma_kernel<<<(tiles + 3) / 4, 128, 0, stream>>>(
                        cur, wh[wbase[sc] + l], nxt, R, curLd, w.cout);
                    cur = nxt; curLd = w.cout;
                    nxt = (nxt == bufA) ? bufB : bufA;
                }
                maxpool_h_kernel<<<(S1 * curLd + 255) / 256, 256, 0, stream>>>(
                    cur, S1, K, curLd, pooled1 + (size_t)b * S1 * 320 + coff, 320);
            }
            coff += (sc == 0) ? 64 : 128;   // outputs 64,128,128 -> 320 total
        }
    }

    // --- SA2 ---
    fps_kernel<<<B, FPS_THREADS, 0, stream>>>(nx1, S1, S2, nx2);
    {
        const float radii[3] = {0.2f, 0.4f, 0.8f};
        const int   ks[3]    = {32, 64, 128};
        const int   wbase[3] = {8, 11, 14};
        int coff = 0;
        for (int sc = 0; sc < 3; ++sc) {
            int K = ks[sc]; float r2 = radii[sc] * radii[sc];
            for (int b = 0; b < B; ++b) {
                const float* xb = nx1 + (size_t)b * S1 * 3;
                const float* cb = nx2 + (size_t)b * S2 * 3;
                ball_query_kernel<<<(S2 + 127) / 128, 128, 0, stream>>>(
                    xb, cb, S1, S2, K, r2, bidx);
                int R = S2 * K, ldx = 352;      // 3 + 320 -> pad 352
                size_t tot = (size_t)R * ldx;
                build_grouped_kernel<<<(int)((tot + 255) / 256), 256, 0, stream>>>(
                    xb, cb, (const float*)nullptr, pooled1 + (size_t)b * S1 * 320,
                    bidx, bufX, S2, K, 320, ldx, 1);
                const _Float16* cur = bufX; int curLd = ldx;
                _Float16* nxt = bufA;
                for (int l = 0; l < 3; ++l) {
                    const WSpec& w = wspec[wbase[sc] + l];
                    int tiles = (R / 16) * (w.cout / 32);
                    gemm_relu_wmma_kernel<<<(tiles + 3) / 4, 128, 0, stream>>>(
                        cur, wh[wbase[sc] + l], nxt, R, curLd, w.cout);
                    cur = nxt; curLd = w.cout;
                    nxt = (nxt == bufA) ? bufB : bufA;
                }
                maxpool_h_kernel<<<(S2 * curLd + 255) / 256, 256, 0, stream>>>(
                    cur, S2, K, curLd, pooled2 + (size_t)b * S2 * 384 + coff, 384);
            }
            coff += 128;                        // 3 x 128 -> 384 total
        }
    }

    // --- SA3 (group-all): rows = B*128, Cin = 3+384 -> pad 416 ---
    for (int b = 0; b < B; ++b) {
        size_t tot = (size_t)S2 * 416;
        build_grouped_kernel<<<(int)((tot + 255) / 256), 256, 0, stream>>>(
            nx2 + (size_t)b * S2 * 3, nx2 + (size_t)b * S2 * 3,
            (const float*)nullptr, pooled2 + (size_t)b * S2 * 384,
            (const int*)nullptr, bufX + (size_t)b * S2 * 416,
            /*S=*/1, /*K=*/S2, /*C=*/384, /*ldx=*/416, /*subtract=*/0);
    }
    {
        int R = B * S2;                         // 1024
        int tiles1 = (R / 16) * (512 / 32);
        gemm_relu_wmma_kernel<<<(tiles1 + 3) / 4, 128, 0, stream>>>(
            bufX, wh[17], bufA, R, 416, 512);
        gemm_relu_wmma_kernel<<<(tiles1 + 3) / 4, 128, 0, stream>>>(
            bufA, wh[18], bufB, R, 512, 512);
        maxpool_f_kernel<<<(B * 512 + 255) / 256, 256, 0, stream>>>(
            bufB, B, S2, 512, (float*)d_out);
    }
}
